// RelationNetwork_15092515078273
// MI455X (gfx1250) — matrix operand order repaired and checked by reading, not compiled, not run on hardware
//
#include <hip/hip_runtime.h>
#include <hip/hip_bf16.h>

typedef __attribute__((ext_vector_type(16))) _Float16 v16h;
typedef __attribute__((ext_vector_type(8)))  _Float16 v8h;
typedef __attribute__((ext_vector_type(8)))  float    v8f;

// ---------------- problem constants ----------------
constexpr int Bc = 128, Tc = 512, LQ = 256, Nc = 256;
constexpr int DN = 100, DH = 300;
constexpr int KN = 128;               // padded node-dim (K for 16x16x32 f16)
constexpr int KH = 320;               // padded hidden dim
constexpr int HC = 640;               // padded concat hidden
constexpr long ROWS = (long)Bc * Tc;  // 65536 path rows

// ---------------- workspace layout (bytes) ----------------
constexpr size_t O_HID16  = 0;
constexpr size_t SZ_HID16 = (size_t)ROWS * KH * 2;            // 41.9 MB
constexpr size_t O_CTX16  = O_HID16 + SZ_HID16;
constexpr size_t SZ_CTX16 = (size_t)Bc * LQ * KH * 2;         // 21.0 MB
constexpr size_t O_ALPHA  = O_CTX16 + SZ_CTX16;
constexpr size_t SZ_ALPHA = (size_t)Bc * Tc * 4;
constexpr size_t O_BMEAN  = O_ALPHA + SZ_ALPHA;
constexpr size_t SZ_BMEAN = (size_t)KN * KH * 2;
constexpr size_t O_BGF    = O_BMEAN + SZ_BMEAN;
constexpr size_t SZ_BG    = (size_t)KH * (3 * KH) * 2;        // gate-packed [r|z|n]
constexpr size_t O_BGB    = O_BGF + SZ_BG;
constexpr size_t O_BO     = O_BGB + SZ_BG;
constexpr size_t SZ_BO    = (size_t)HC * KH * 2;
constexpr size_t O_BN     = O_BO + SZ_BO;

// output element offsets
constexpr long OUT_PATH = 0;
constexpr long OUT_NODE = (long)Bc * DH;                      // 38400
constexpr long OUT_MASK = OUT_NODE + (long)Bc * Nc * DH;      // 9868800

// ---------------- WMMA helpers ----------------
static __device__ __forceinline__ v8f wmma_f16(v8f c, v16h a, v16h b) {
  // D = A(16x32 f16) * B(32x16 f16) + C(16x16 f32)
  return __builtin_amdgcn_wmma_f32_16x16x32_f16(false, a, false, b, (short)0, c,
                                                false, false);
}

static __device__ __forceinline__ v8f zero8() {
  v8f c = {0.f, 0.f, 0.f, 0.f, 0.f, 0.f, 0.f, 0.f};
  return c;
}

// A fragment (16x32 f16) from a row-major [rows][stride] half array.
// Lane l holds row m=l&15; halves 0..15 -> K=hi*16+i.
static __device__ __forceinline__ v16h load_a_frag(const _Float16* base, int stride,
                                                   int kbase, int lane) {
  int m = lane & 15, hi = lane >> 4;
  const _Float16* p = base + (long)m * stride + kbase + hi * 16;
  union { v16h v; v8h h[2]; } f;
  f.h[0] = *(const v8h*)p;
  f.h[1] = *(const v8h*)(p + 8);
  return f.v;
}

// B fragment from fragment-linear packed weights: tile-major, 32 lanes x 16 halves.
static __device__ __forceinline__ v16h load_b_packed(const _Float16* base, int tile,
                                                     int lane) {
  return *(const v16h*)(base + ((long)tile * 32 + lane) * 16);
}

static __device__ __forceinline__ float sigmoidf_(float x) {
  return 1.0f / (1.0f + __expf(-x));
}

// ---------------- prep: pack weights (f16, B-fragment-linear) ----------------
// dst element idx = tile*512 + lane*16 + j ; maps to B[kt*32+(lane>>4)*16+j][nt*16+(lane&15)]
__global__ void k_pack_b(const float* __restrict__ src, _Float16* __restrict__ dst,
                         int KT, int NT, int mode) {
  int idx = blockIdx.x * 256 + threadIdx.x;
  int total = KT * NT * 512;
  if (idx >= total) return;
  int j    = idx & 15;
  int lane = (idx >> 4) & 31;
  int tile = idx >> 9;
  int nt = tile % NT, kt = tile / NT;
  int k = kt * 32 + ((lane >> 4) << 4) + j;
  int n = nt * 16 + (lane & 15);
  float v = 0.f;
  if (mode == 0) {              // Linear(100,300): B[k][n] = W[n*100+k]
    if (k < DN && n < DH) v = src[n * DN + k];
  } else if (mode == 1) {       // GRU W_ih [900,300], gate-packed cols [r|z|n]x320
    int g = n / KH, u = n % KH;
    if (u < DH && k < DH) v = src[(g * DH + u) * DH + k];
  } else {                      // W_out [300,600], K = [h_f pad 320 | h_b pad 320]
    int seg = k / KH, ku = k % KH;
    if (ku < DH && n < DH) v = src[n * (2 * DH) + seg * DH + ku];
  }
  dst[idx] = (_Float16)v;
}

// context f32 -> padded f16 rows [B*LQ][KH]
__global__ void k_prep_ctx(const float* __restrict__ ctx, _Float16* __restrict__ dst) {
  long idx = (long)blockIdx.x * 256 + threadIdx.x;
  long total = (long)Bc * LQ * KH;
  if (idx >= total) return;
  int d = (int)(idx % KH);
  long row = idx / KH;
  dst[idx] = (_Float16)((d < DH) ? ctx[row * DH + d] : 0.f);
}

__global__ void k_zero_f32(float* __restrict__ p, int n) {
  int i = blockIdx.x * 256 + threadIdx.x;
  if (i < n) p[i] = 0.f;
}

// ---------------- main path chain: gather -> mean -> biGRU -> out ----------------
// One wave per workgroup, 32 rows (M-blocked x2): every B fragment feeds 2 WMMAs,
// halving L2 weight traffic. LDS: xA 20KB + scratch 40KB (triA overlaid on hidA).
__global__ __launch_bounds__(32) void k_main(
    const float* __restrict__ nodes, const int* __restrict__ trip,
    const float* __restrict__ b_mean,
    const float* __restrict__ b_ih_f, const float* __restrict__ b_hh_f,
    const float* __restrict__ b_ih_b, const float* __restrict__ b_hh_b,
    const float* __restrict__ b_out,
    const _Float16* __restrict__ Bmean, const _Float16* __restrict__ Bgf,
    const _Float16* __restrict__ Bgb, const _Float16* __restrict__ Bo,
    _Float16* __restrict__ hid16) {
  __shared__ _Float16 xA[32 * KH];       // 20 KB
  __shared__ _Float16 scratch[32 * HC];  // 40 KB: triA (gather) then hidA (GRU out)
  _Float16* triA = scratch;              // [32][KN]
  _Float16* hidA = scratch;              // [32][HC]

  const int lane = threadIdx.x;
  const int hi = lane >> 4, ncol = lane & 15;
  const long tileRow = blockIdx.x;       // 0..2047, 32 rows each

  // ---- gather head/tail nodes, tri_mean, pad to KN (one row per lane) ----
  {
    long R = tileRow * 32 + lane;
    int bi = (int)(R >> 9);
    int t  = (int)(R & 511);
    int h  = trip[((long)bi * Tc + t) * 3 + 0];
    int tl = trip[((long)bi * Tc + t) * 3 + 2];
    const float* ph = nodes + ((long)bi * Nc + h) * DN;
    const float* pt = nodes + ((long)bi * Nc + tl) * DN;
    for (int k = 0; k < KN; ++k) {
      float v = (k < DN) ? 0.5f * (ph[k] + pt[k]) : 0.f;
      triA[lane * KN + k] = (_Float16)v;
    }
  }
  __syncthreads();

  // ---- x = tri_mean @ W_mean^T + b_mean  (pad cols -> 0) ----
  for (int nt = 0; nt < KH / 16; ++nt) {
    v8f c0 = zero8(), c1 = zero8();
    for (int kt = 0; kt < KN / 32; ++kt) {
      v16h b  = load_b_packed(Bmean, kt * (KH / 16) + nt, lane);
      v16h a0 = load_a_frag(triA, KN, kt * 32, lane);
      v16h a1 = load_a_frag(triA + 16 * KN, KN, kt * 32, lane);
      c0 = wmma_f16(c0, a0, b);
      c1 = wmma_f16(c1, a1, b);
    }
    int u = nt * 16 + ncol;
    float bias = (u < DH) ? b_mean[u] : 0.f;
#pragma unroll
    for (int vi = 0; vi < 8; ++vi) {
      int m = vi + 8 * hi;
      xA[m * KH + u]        = (_Float16)((u < DH) ? (c0[vi] + bias) : 0.f);
      xA[(16 + m) * KH + u] = (_Float16)((u < DH) ? (c1[vi] + bias) : 0.f);
    }
  }
  __syncthreads();  // triA fully consumed; scratch becomes hidA

  // ---- bidirectional single-step GRU (h0=0); gates fused per 16-col u-tile ----
  for (int dir = 0; dir < 2; ++dir) {
    const _Float16* Bg = dir ? Bgb : Bgf;
    const float* bih = dir ? b_ih_b : b_ih_f;
    const float* bhh = dir ? b_hh_b : b_hh_f;
    for (int ut = 0; ut < KH / 16; ++ut) {
      v8f cr0 = zero8(), cz0 = zero8(), cn0 = zero8();
      v8f cr1 = zero8(), cz1 = zero8(), cn1 = zero8();
      for (int kt = 0; kt < KH / 32; ++kt) {
        v16h a0 = load_a_frag(xA, KH, kt * 32, lane);
        v16h a1 = load_a_frag(xA + 16 * KH, KH, kt * 32, lane);
        v16h br = load_b_packed(Bg, kt * 60 + ut, lane);
        v16h bz = load_b_packed(Bg, kt * 60 + 20 + ut, lane);
        v16h bn = load_b_packed(Bg, kt * 60 + 40 + ut, lane);
        cr0 = wmma_f16(cr0, a0, br);
        cr1 = wmma_f16(cr1, a1, br);
        cz0 = wmma_f16(cz0, a0, bz);
        cz1 = wmma_f16(cz1, a1, bz);
        cn0 = wmma_f16(cn0, a0, bn);
        cn1 = wmma_f16(cn1, a1, bn);
      }
      int u = ut * 16 + ncol;
      bool valid = u < DH;
      float bir = valid ? bih[u] : 0.f,           bhr = valid ? bhh[u] : 0.f;
      float biz = valid ? bih[DH + u] : 0.f,      bhz = valid ? bhh[DH + u] : 0.f;
      float bin_ = valid ? bih[2 * DH + u] : 0.f, bhn = valid ? bhh[2 * DH + u] : 0.f;
#pragma unroll
      for (int vi = 0; vi < 8; ++vi) {
        int m = vi + 8 * hi;
        float rr = sigmoidf_(cr0[vi] + bir + bhr);
        float zz = sigmoidf_(cz0[vi] + biz + bhz);
        float nn = tanhf(cn0[vi] + bin_ + rr * bhn);
        hidA[m * HC + dir * KH + u] =
            (_Float16)(valid ? (1.f - zz) * nn : 0.f);
        rr = sigmoidf_(cr1[vi] + bir + bhr);
        zz = sigmoidf_(cz1[vi] + biz + bhz);
        nn = tanhf(cn1[vi] + bin_ + rr * bhn);
        hidA[(16 + m) * HC + dir * KH + u] =
            (_Float16)(valid ? (1.f - zz) * nn : 0.f);
      }
    }
  }
  __syncthreads();

  // ---- hid = tanh([h_f,h_b] @ W_out^T + b_out), write padded f16 rows ----
  for (int nt = 0; nt < KH / 16; ++nt) {
    v8f c0 = zero8(), c1 = zero8();
    for (int kt = 0; kt < HC / 32; ++kt) {
      v16h b  = load_b_packed(Bo, kt * (KH / 16) + nt, lane);
      v16h a0 = load_a_frag(hidA, HC, kt * 32, lane);
      v16h a1 = load_a_frag(hidA + 16 * HC, HC, kt * 32, lane);
      c0 = wmma_f16(c0, a0, b);
      c1 = wmma_f16(c1, a1, b);
    }
    int u = nt * 16 + ncol;
    float bias = (u < DH) ? b_out[u] : 0.f;
#pragma unroll
    for (int vi = 0; vi < 8; ++vi) {
      int m = vi + 8 * hi;
      long row0 = tileRow * 32 + m;
      hid16[row0 * KH + u] =
          (_Float16)((u < DH) ? tanhf(c0[vi] + bias) : 0.f);
      hid16[(row0 + 16) * KH + u] =
          (_Float16)((u < DH) ? tanhf(c1[vi] + bias) : 0.f);
    }
  }
}

// ---------------- attention: alpha[b,t] = sum_l softmax_t(ctx@hid^T) ----------------
__global__ __launch_bounds__(32) void k_attn(const _Float16* __restrict__ ctx16,
                                             const _Float16* __restrict__ hid16,
                                             float* __restrict__ alpha) {
  const int lane = threadIdx.x;
  const int bi = blockIdx.x >> 4;
  const int lt = blockIdx.x & 15;
  const int ncol = lane & 15, hi = lane >> 4;
  const _Float16* A = ctx16 + ((long)bi * LQ + lt * 16) * KH;
  const _Float16* H = hid16 + (long)bi * Tc * KH;

  // hoist the 10 A-fragments (queries are reused across all 32 t-tiles, 2 passes)
  v16h af[KH / 32];
#pragma unroll
  for (int kt = 0; kt < KH / 32; ++kt) af[kt] = load_a_frag(A, KH, kt * 32, lane);

  float mrow[8], srow[8];
#pragma unroll
  for (int vi = 0; vi < 8; ++vi) { mrow[vi] = -1e30f; srow[vi] = 0.f; }

  for (int pass = 0; pass < 2; ++pass) {
    for (int tt = 0; tt < Tc / 16; ++tt) {
      v8f c = zero8();
      for (int kt = 0; kt < KH / 32; ++kt) {
        const _Float16* p = H + (long)(tt * 16 + ncol) * KH + kt * 32 + hi * 16;
        v16h b = *(const v16h*)p;
        c = wmma_f16(c, af[kt], b);
      }
      if (pass == 0) {  // online row max + sumexp (rows live in 16-lane halves)
#pragma unroll
        for (int vi = 0; vi < 8; ++vi) {
          float tmax = c[vi];
          for (int d = 1; d < 16; d <<= 1) tmax = fmaxf(tmax, __shfl_xor(tmax, d));
          float mnew = fmaxf(mrow[vi], tmax);
          float pe = __expf(c[vi] - mnew);
          float rsum = pe;
          for (int d = 1; d < 16; d <<= 1) rsum += __shfl_xor(rsum, d);
          srow[vi] = srow[vi] * __expf(mrow[vi] - mnew) + rsum;
          mrow[vi] = mnew;
        }
      } else {          // accumulate alpha[t] += sum_l p(l,t)
        float asum = 0.f;
#pragma unroll
        for (int vi = 0; vi < 8; ++vi) asum += __expf(c[vi] - mrow[vi]) / srow[vi];
        asum += __shfl_xor(asum, 16);
        if (lane < 16) atomicAdd(&alpha[(long)bi * Tc + tt * 16 + ncol], asum);
      }
    }
  }
}

// ---------------- rep = alpha @ hid ; layernorm -> path_feature ----------------
__global__ __launch_bounds__(512) void k_repln(const float* __restrict__ alpha,
                                               const _Float16* __restrict__ hid16,
                                               const float* __restrict__ gamma,
                                               const float* __restrict__ beta,
                                               float* __restrict__ out) {
  __shared__ float alphaS[Tc];
  __shared__ float red[512];
  const int bi = blockIdx.x, tid = threadIdx.x;
  alphaS[tid] = alpha[(long)bi * Tc + tid];
  __syncthreads();

  float acc = 0.f;
  if (tid < DH) {
    const _Float16* Hp = hid16 + (long)bi * Tc * KH + tid;
    for (int t = 0; t < Tc; ++t) acc += alphaS[t] * (float)Hp[(long)t * KH];
  }
  red[tid] = (tid < DH) ? acc : 0.f;
  __syncthreads();
  for (int s = 256; s > 0; s >>= 1) {
    if (tid < s) red[tid] += red[tid + s];
    __syncthreads();
  }
  float mean = red[0] / (float)DH;
  __syncthreads();
  float dv = (tid < DH) ? (acc - mean) : 0.f;
  red[tid] = dv * dv;
  __syncthreads();
  for (int s = 256; s > 0; s >>= 1) {
    if (tid < s) red[tid] += red[tid + s];
    __syncthreads();
  }
  float var = red[0] / (float)DH;
  if (tid < DH)
    out[(long)bi * DH + tid] =
        (acc - mean) * rsqrtf(var + 1e-5f) * gamma[tid] + beta[tid];
}

// ---------------- node_feature = nodes @ W_node^T + b_node ----------------
__global__ __launch_bounds__(32) void k_node(const float* __restrict__ nodes,
                                             const _Float16* __restrict__ Bn,
                                             const float* __restrict__ b_node,
                                             float* __restrict__ outNode) {
  __shared__ _Float16 nA[16 * KN];
  const int lane = threadIdx.x;
  const int r = lane & 15, hi = lane >> 4, ncol = lane & 15;
  const long row0 = (long)blockIdx.x * 16;

  const float* src = nodes + (row0 + r) * DN;
  for (int k = hi * 64; k < hi * 64 + 64; ++k)
    nA[r * KN + k] = (_Float16)((k < DN) ? src[k] : 0.f);
  __syncthreads();

  v16h af[KN / 32];
#pragma unroll
  for (int kt = 0; kt < KN / 32; ++kt) af[kt] = load_a_frag(nA, KN, kt * 32, lane);

  for (int nt = 0; nt < KH / 16; ++nt) {
    v8f c = zero8();
#pragma unroll
    for (int kt = 0; kt < KN / 32; ++kt)
      c = wmma_f16(c, af[kt], load_b_packed(Bn, kt * (KH / 16) + nt, lane));
    int u = nt * 16 + ncol;
    if (u < DH) {
      float bias = b_node[u];
#pragma unroll
      for (int vi = 0; vi < 8; ++vi)
        outNode[(row0 + vi + 8 * hi) * DH + u] = c[vi] + bias;
    }
  }
}

// ---------------- launcher ----------------
extern "C" void kernel_launch(void* const* d_in, const int* in_sizes, int n_in,
                              void* d_out, int out_size, void* d_ws, size_t ws_size,
                              hipStream_t stream) {
  (void)in_sizes; (void)n_in; (void)out_size; (void)ws_size;
  const float* nodes  = (const float*)d_in[0];
  const float* ctx    = (const float*)d_in[1];
  const int*   trip   = (const int*)d_in[2];
  const float* W_mean = (const float*)d_in[4];
  const float* b_mean = (const float*)d_in[5];
  const float* W_ih_f = (const float*)d_in[6];
  const float* b_ih_f = (const float*)d_in[7];
  const float* b_hh_f = (const float*)d_in[8];
  const float* W_ih_b = (const float*)d_in[9];
  const float* b_ih_b = (const float*)d_in[10];
  const float* b_hh_b = (const float*)d_in[11];
  const float* W_out  = (const float*)d_in[12];
  const float* b_out  = (const float*)d_in[13];
  const float* W_node = (const float*)d_in[14];
  const float* b_node = (const float*)d_in[15];
  const float* gamma  = (const float*)d_in[16];
  const float* beta   = (const float*)d_in[17];
  float* out = (float*)d_out;

  char* ws = (char*)d_ws;
  _Float16* hid16 = (_Float16*)(ws + O_HID16);
  _Float16* ctx16 = (_Float16*)(ws + O_CTX16);
  float*    alpha = (float*)(ws + O_ALPHA);
  _Float16* Bmean = (_Float16*)(ws + O_BMEAN);
  _Float16* Bgf   = (_Float16*)(ws + O_BGF);
  _Float16* Bgb   = (_Float16*)(ws + O_BGB);
  _Float16* Bo    = (_Float16*)(ws + O_BO);
  _Float16* Bn    = (_Float16*)(ws + O_BN);

  // pack weights to f16 fragment-linear layout
  k_pack_b<<<dim3((4 * 20 * 512 + 255) / 256), 256, 0, stream>>>(W_mean, Bmean, 4, 20, 0);
  k_pack_b<<<dim3((10 * 60 * 512 + 255) / 256), 256, 0, stream>>>(W_ih_f, Bgf, 10, 60, 1);
  k_pack_b<<<dim3((10 * 60 * 512 + 255) / 256), 256, 0, stream>>>(W_ih_b, Bgb, 10, 60, 1);
  k_pack_b<<<dim3((20 * 20 * 512 + 255) / 256), 256, 0, stream>>>(W_out, Bo, 20, 20, 2);
  k_pack_b<<<dim3((4 * 20 * 512 + 255) / 256), 256, 0, stream>>>(W_node, Bn, 4, 20, 0);

  // context -> padded f16; zero alpha; zero node_mask output
  long ctxTotal = (long)Bc * LQ * KH;
  k_prep_ctx<<<dim3((unsigned)((ctxTotal + 255) / 256)), 256, 0, stream>>>(ctx, ctx16);
  k_zero_f32<<<dim3((Bc * Tc + 255) / 256), 256, 0, stream>>>(alpha, Bc * Tc);
  k_zero_f32<<<dim3((Bc * Nc + 255) / 256), 256, 0, stream>>>(out + OUT_MASK, Bc * Nc);

  // path chain: 2048 single-wave workgroups, 32 rows each (M-blocked x2)
  k_main<<<dim3((unsigned)(ROWS / 32)), 32, 0, stream>>>(
      nodes, trip, b_mean, b_ih_f, b_hh_f, b_ih_b, b_hh_b, b_out,
      Bmean, Bgf, Bgb, Bo, hid16);

  // attention: (b, 16-query-row tile) per wave
  k_attn<<<dim3(Bc * (LQ / 16)), 32, 0, stream>>>(ctx16, hid16, alpha);

  // rep + layernorm -> path_feature
  k_repln<<<dim3(Bc), 512, 0, stream>>>(alpha, hid16, gamma, beta, out + OUT_PATH);

  // node projection -> node_feature
  k_node<<<dim3((unsigned)((long)Bc * Nc / 16)), 32, 0, stream>>>(
      nodes, Bn, b_node, out + OUT_NODE);
}
